// LocalSample_5815385719334
// MI455X (gfx1250) — compile-verified, matrix operand order checked by reading, not run on hardware
//
#include <hip/hip_runtime.h>

// ---------------------------------------------------------------------------
// Problem constants (from the reference)
// ---------------------------------------------------------------------------
#define B_    8
#define N_    577
#define NH    12
#define DK    64
#define D_    768
#define FF    3072
#define S_    145          // 1 + NH * KSEL selected tokens
#define SP    160          // S_ padded to multiple of 32 (WMMA K granularity)
#define MROWS (B_ * S_)    // 1160 total token rows
#define KSEL  12
#define NEG_BIG (-3.402823e38f)

typedef __attribute__((ext_vector_type(16))) _Float16     v16h;
typedef __attribute__((ext_vector_type(8)))  float        v8f;
typedef __attribute__((ext_vector_type(4)))  unsigned int v4u;
typedef __attribute__((ext_vector_type(8)))  int          v8i;
typedef __attribute__((ext_vector_type(4)))  int          v4i;

#if defined(__has_builtin)
#if __has_builtin(__builtin_amdgcn_tensor_load_to_lds) && \
    __has_builtin(__builtin_amdgcn_s_wait_tensorcnt)
#define USE_TDM 1
#endif
#endif
#ifndef USE_TDM
#define USE_TDM 0
#endif

static __device__ __forceinline__ int imin(int a, int b) { return a < b ? a : b; }

// Load one WMMA 16-bit A/B fragment (K-contiguous operand, slab-relative):
//   lanes 0-15 : halves 0..7 = K +0..7,   halves 8..15 = K +16..23
//   lanes 16-31: halves 0..7 = K +8..15,  halves 8..15 = K +24..31
// caller passes p = rowBase + k0 + ((lane>>4)<<3); works for global or LDS.
static __device__ __forceinline__ v16h load_frag_k(const _Float16* __restrict__ p) {
  v16h r;
#pragma unroll
  for (int i = 0; i < 8; ++i) { r[i] = p[i]; r[8 + i] = p[16 + i]; }
  return r;
}

static __device__ __forceinline__ v8f wmma_f16(v16h a, v16h b, v8f c) {
  return __builtin_amdgcn_wmma_f32_16x16x32_f16(false, a, false, b, (short)0, c,
                                                false, false);
}

#if USE_TDM
// Issue a TDM load of a (64 rows x 32 halves) tile of Bt[N][K] into LDS.
// Descriptor per CDNA5 ISA 8.3/8.4: group0 = {count/lds/global/type},
// group1 = {data_size, tensor dims, tile dims, dim0 stride}.
static __device__ __forceinline__ void tdm_load_b_tile(const _Float16* gsrc,
                                                       unsigned lds_base,
                                                       int K, int N, int ldb) {
  unsigned long long gaddr = (unsigned long long)(const void*)gsrc;
  v4u g0;
  g0[0] = 1u;                                             // count=1, no gather
  g0[1] = lds_base;                                       // lds_addr (bytes)
  g0[2] = (unsigned)(gaddr & 0xffffffffu);                // global_addr[31:0]
  g0[3] = (unsigned)((gaddr >> 32) & 0x01ffffffu)         // global_addr[56:32]
        | (2u << 30);                                     // type = 2 ("image")
  v8i g1;
  g1[0] = (int)(1u << 16);                                // mask=0, data_size=2B
  g1[1] = (int)(((unsigned)K & 0xffffu) << 16);           // tensor_dim0 lo16
  g1[2] = (int)((((unsigned)K >> 16) & 0xffffu)           // tensor_dim0 hi16
        | (((unsigned)N & 0xffffu) << 16));               // tensor_dim1 lo16
  g1[3] = (int)((((unsigned)N >> 16) & 0xffffu)           // tensor_dim1 hi16
        | (32u << 16));                                   // tile_dim0 = 32 elems
  g1[4] = (int)64u;                                       // tile_dim1=64, dim2=0
  g1[5] = (int)ldb;                                       // dim0_stride lo32
  g1[6] = 0;                                              // stride hi16, d1s lo16
  g1[7] = 0;                                              // d1s hi32
  v4i gz; gz[0] = 0; gz[1] = 0; gz[2] = 0; gz[3] = 0;     // 2D: groups 2/3 unused
#if defined(__clang_major__) && __clang_major__ >= 23
  v8i gz8;
#pragma unroll
  for (int q = 0; q < 8; ++q) gz8[q] = 0;
  __builtin_amdgcn_tensor_load_to_lds(g0, g1, gz, gz, gz8, 0);
#else
  __builtin_amdgcn_tensor_load_to_lds(g0, g1, gz, gz, 0);
#endif
}
#endif

// ---------------------------------------------------------------------------
// Weight convert + transpose: Wt[n*K + k] = (f16) W[k*N + n]
// ---------------------------------------------------------------------------
__global__ void convert_transpose_kernel(const float* __restrict__ W,
                                         _Float16* __restrict__ Wt,
                                         int K, int N) {
  long i = (long)blockIdx.x * blockDim.x + threadIdx.x;
  long total = (long)K * N;
  if (i >= total) return;
  int n = (int)(i / K);
  int k = (int)(i % K);
  Wt[i] = (_Float16)W[(long)k * N + n];
}

// ---------------------------------------------------------------------------
// Attention rollout step: vout[bh][m] = sum_n vin[bh][n] * A[bh][n][m]
// Bandwidth bound (reads one full 577x577 layer per bh); vector held in LDS.
// ---------------------------------------------------------------------------
__global__ void rollout_kernel(const float* __restrict__ vin, long vin_stride,
                               const float* __restrict__ A,
                               float* __restrict__ vout) {
  __shared__ float lv[N_];
  const int bh = blockIdx.y;
  const float* vi = vin + (long)bh * vin_stride;
  for (int i = threadIdx.x; i < N_; i += blockDim.x) lv[i] = vi[i];
  __syncthreads();
  const int m = blockIdx.x * blockDim.x + threadIdx.x;
  if (m >= N_) return;
  const float* Ab = A + (long)bh * N_ * N_;
  float acc = 0.f;
  for (int n = 0; n < N_; ++n) acc = fmaf(lv[n], Ab[(long)n * N_ + m], acc);
  vout[(long)bh * N_ + m] = acc;
}

// ---------------------------------------------------------------------------
// top-k (k=12) over last_map = v[bh][1..576]; jax tie rule: ties -> lower idx.
// ---------------------------------------------------------------------------
__global__ void topk_kernel(const float* __restrict__ vfin, int* __restrict__ idx) {
  __shared__ float vals[576];
  __shared__ float rv[192];
  __shared__ int   ri[192];
  const int bh = blockIdx.x;
  const int b = bh / NH, h = bh % NH;
  const float* src = vfin + (long)bh * N_ + 1;
  const int tid = threadIdx.x;               // blockDim = 192
  for (int j = tid; j < 576; j += 192) vals[j] = src[j];
  __syncthreads();
  for (int r = 0; r < KSEL; ++r) {
    float bvv = NEG_BIG; int bii = 1 << 30;
    for (int j = tid; j < 576; j += 192) {
      float vv = vals[j];
      if (vv > bvv || (vv == bvv && j < bii)) { bvv = vv; bii = j; }
    }
    rv[tid] = bvv; ri[tid] = bii;
    __syncthreads();
    if (tid < 64) {
      int o = tid + 128;
      if (rv[o] > rv[tid] || (rv[o] == rv[tid] && ri[o] < ri[tid])) { rv[tid] = rv[o]; ri[tid] = ri[o]; }
    }
    __syncthreads();
    for (int s = 64; s > 0; s >>= 1) {
      if (tid < s) {
        int o = tid + s;
        if (rv[o] > rv[tid] || (rv[o] == rv[tid] && ri[o] < ri[tid])) { rv[tid] = rv[o]; ri[tid] = ri[o]; }
      }
      __syncthreads();
    }
    if (tid == 0) {
      idx[b * (NH * KSEL) + h * KSEL + r] = ri[0] + 1;
      vals[ri[0]] = NEG_BIG;
    }
    __syncthreads();
  }
}

// ---------------------------------------------------------------------------
// Gather selected tokens, LayerNorm1 (ddof=1, denom = sd+eps as in reference).
// ---------------------------------------------------------------------------
__global__ void gather_ln_kernel(const float* __restrict__ x, const int* __restrict__ idx,
                                 const float* __restrict__ g, const float* __restrict__ bb,
                                 float* __restrict__ fl, _Float16* __restrict__ n1h) {
  __shared__ float red[256];
  const int row = blockIdx.x;                // B_*S_ rows
  const int b = row / S_, s = row % S_;
  const int tok = (s == 0) ? 0 : idx[b * (NH * KSEL) + (s - 1)];
  const float* src = x + ((long)b * N_ + tok) * D_;
  const int tid = threadIdx.x;               // 256 threads, 3 elems each
  float v0 = src[tid], v1 = src[tid + 256], v2 = src[tid + 512];
  red[tid] = v0 + v1 + v2;
  __syncthreads();
  for (int s2 = 128; s2 > 0; s2 >>= 1) { if (tid < s2) red[tid] += red[tid + s2]; __syncthreads(); }
  const float mu = red[0] / (float)D_;
  __syncthreads();
  float d0 = v0 - mu, d1 = v1 - mu, d2 = v2 - mu;
  red[tid] = d0 * d0 + d1 * d1 + d2 * d2;
  __syncthreads();
  for (int s2 = 128; s2 > 0; s2 >>= 1) { if (tid < s2) red[tid] += red[tid + s2]; __syncthreads(); }
  const float sd = sqrtf(red[0] / (float)(D_ - 1));
  const float inv = 1.f / (sd + 1e-6f);
  long base = (long)row * D_;
  fl[base + tid] = v0;  fl[base + tid + 256] = v1;  fl[base + tid + 512] = v2;
  n1h[base + tid]       = (_Float16)(g[tid] * d0 * inv + bb[tid]);
  n1h[base + tid + 256] = (_Float16)(g[tid + 256] * d1 * inv + bb[tid + 256]);
  n1h[base + tid + 512] = (_Float16)(g[tid + 512] * d2 * inv + bb[tid + 512]);
}

// LayerNorm2 over f32 rows -> f16 output.
__global__ void ln_kernel(const float* __restrict__ z,
                          const float* __restrict__ g, const float* __restrict__ bb,
                          _Float16* __restrict__ outh) {
  __shared__ float red[256];
  const int row = blockIdx.x;
  const float* src = z + (long)row * D_;
  const int tid = threadIdx.x;
  float v0 = src[tid], v1 = src[tid + 256], v2 = src[tid + 512];
  red[tid] = v0 + v1 + v2;
  __syncthreads();
  for (int s2 = 128; s2 > 0; s2 >>= 1) { if (tid < s2) red[tid] += red[tid + s2]; __syncthreads(); }
  const float mu = red[0] / (float)D_;
  __syncthreads();
  float d0 = v0 - mu, d1 = v1 - mu, d2 = v2 - mu;
  red[tid] = d0 * d0 + d1 * d1 + d2 * d2;
  __syncthreads();
  for (int s2 = 128; s2 > 0; s2 >>= 1) { if (tid < s2) red[tid] += red[tid + s2]; __syncthreads(); }
  const float sd = sqrtf(red[0] / (float)(D_ - 1));
  const float inv = 1.f / (sd + 1e-6f);
  long base = (long)row * D_;
  outh[base + tid]       = (_Float16)(g[tid] * d0 * inv + bb[tid]);
  outh[base + tid + 256] = (_Float16)(g[tid + 256] * d1 * inv + bb[tid + 256]);
  outh[base + tid + 512] = (_Float16)(g[tid + 512] * d2 * inv + bb[tid + 512]);
}

// ---------------------------------------------------------------------------
// Generic WMMA GEMM:  C[M,N] = A[M,K] * Bt[N,K]^T + bias (+resid) (relu?)
// Block = 8 waves = 128 M-rows x 64 N-cols. The 64x32 B slab is staged into
// LDS once per K-step by the Tensor Data Mover (wave 0 issues TENSOR_LOAD_TO_LDS
// and waits TENSORcnt); all 8 waves consume it via ds_load fragments, so the
// weight tile is read from global once per workgroup instead of once per wave.
// N must be a multiple of 64 and K a multiple of 32 (true for all call sites).
// ---------------------------------------------------------------------------
__global__ void gemm_bt_wmma(const _Float16* __restrict__ A, int lda,
                             const _Float16* __restrict__ Bt, int ldb,
                             const float* __restrict__ bias,
                             const float* __restrict__ resid,
                             float* __restrict__ Cf, _Float16* __restrict__ Ch,
                             int M, int N, int K, int ldc, int relu) {
  const int lane = threadIdx.x & 31;
  const int wid = threadIdx.x >> 5;          // 0..7
  const int tm8 = (M + 127) >> 7;            // 128-row blocks
  const int mo = (blockIdx.x % tm8) * 128 + wid * 16;
  const int nb = (blockIdx.x / tm8) * 64;

  const int arow = imin(mo + (lane & 15), M - 1);
  const int kh = (lane >> 4) << 3;
  v8f acc[4] = {};

#if USE_TDM
  __shared__ _Float16 bsm[64 * 32];          // 64 N-rows x 32 K-halves slab
  const unsigned lds_base = (unsigned)(unsigned long long)(const void*)bsm;
  for (int k0 = 0; k0 < K; k0 += 32) {
    if (wid == 0) {
      tdm_load_b_tile(Bt + (long)nb * ldb + k0, lds_base, K, N, ldb);
      __builtin_amdgcn_s_wait_tensorcnt(0);
    }
    __syncthreads();                         // slab visible to all waves
    v16h a = load_frag_k(A + (long)arow * lda + k0 + kh);
#pragma unroll
    for (int t = 0; t < 4; ++t) {
      v16h bf = load_frag_k(bsm + (t * 16 + (lane & 15)) * 32 + kh);
      acc[t] = wmma_f16(a, bf, acc[t]);
    }
    __syncthreads();                         // done before next slab overwrite
  }
#else
  int nrow[4];
#pragma unroll
  for (int t = 0; t < 4; ++t) nrow[t] = nb + t * 16 + (lane & 15);
  for (int k0 = 0; k0 < K; k0 += 32) {
    v16h a = load_frag_k(A + (long)arow * lda + k0 + kh);
#pragma unroll
    for (int t = 0; t < 4; ++t) {
      v16h bf = load_frag_k(Bt + (long)nrow[t] * ldb + k0 + kh);
      acc[t] = wmma_f16(a, bf, acc[t]);
    }
  }
#endif

  const int mbase = mo + ((lane >> 4) << 3);
  const int ncol = lane & 15;
#pragma unroll
  for (int r = 0; r < 8; ++r) {
    const int m = mbase + r;
    if (m >= M) continue;
#pragma unroll
    for (int t = 0; t < 4; ++t) {
      const int n = nb + t * 16 + ncol;
      float val = acc[t][r] + bias[n];
      if (resid) val += resid[(long)m * ldc + n];
      if (relu) val = fmaxf(val, 0.f);
      if (Cf) Cf[(long)m * ldc + n] = val;
      else    Ch[(long)m * ldc + n] = (_Float16)val;
    }
  }
}

// ---------------------------------------------------------------------------
// Batched score GEMM per (b,h): sc = (Q Kt^T) / 8, f32, padded SPxSP rows.
// ---------------------------------------------------------------------------
__global__ void scores_wmma(const _Float16* __restrict__ q,
                            const _Float16* __restrict__ kmat,
                            float* __restrict__ sc) {
  const int lane = threadIdx.x;
  const int bh = blockIdx.y, b = bh / NH, h = bh % NH;
  const int t = blockIdx.x;                  // 10 x 10 tiles
  const int mo = (t % 10) << 4, no = (t / 10) << 4;
  const _Float16* qb = q + (long)b * S_ * D_ + h * DK;
  const _Float16* kb = kmat + (long)b * S_ * D_ + h * DK;
  const int arow = imin(mo + (lane & 15), S_ - 1);
  const int brow = imin(no + (lane & 15), S_ - 1);
  const int kh = (lane >> 4) << 3;
  v8f acc = {};
  for (int k0 = 0; k0 < DK; k0 += 32) {
    v16h a = load_frag_k(qb + (long)arow * D_ + k0 + kh);
    v16h bf = load_frag_k(kb + (long)brow * D_ + k0 + kh);
    acc = wmma_f16(a, bf, acc);
  }
  const int mbase = mo + ((lane >> 4) << 3);
  const int n = no + (lane & 15);
  if (n < S_) {
#pragma unroll
    for (int r = 0; r < 8; ++r) {
      const int m = mbase + r;
      if (m < S_) sc[((long)bh * SP + m) * SP + n] = acc[r] * 0.125f;
    }
  }
}

// ---------------------------------------------------------------------------
// Dual softmax: pa = softmax(-sc), pc = softmax(sc); f16 out, zero K-pad to SP.
// ---------------------------------------------------------------------------
__global__ void softmax_pm_kernel(const float* __restrict__ sc,
                                  _Float16* __restrict__ pa,
                                  _Float16* __restrict__ pc) {
  __shared__ float red[SP];
  const int row = blockIdx.x;                // BH * S_ rows
  const int bh = row / S_, i = row % S_;
  const float* srow = sc + ((long)bh * SP + i) * SP;
  const int tid = threadIdx.x;               // SP threads
  const bool act = tid < S_;
  const float raw = act ? srow[tid] : 0.f;

  red[tid] = act ? raw : NEG_BIG;
  __syncthreads();
  if (tid < 32) red[tid] = fmaxf(red[tid], red[tid + 128]);
  __syncthreads();
  for (int s = 64; s > 0; s >>= 1) { if (tid < s) red[tid] = fmaxf(red[tid], red[tid + s]); __syncthreads(); }
  const float mxp = red[0];
  __syncthreads();
  red[tid] = act ? -raw : NEG_BIG;
  __syncthreads();
  if (tid < 32) red[tid] = fmaxf(red[tid], red[tid + 128]);
  __syncthreads();
  for (int s = 64; s > 0; s >>= 1) { if (tid < s) red[tid] = fmaxf(red[tid], red[tid + s]); __syncthreads(); }
  const float mxn = red[0];
  __syncthreads();

  const float ep = act ? expf(raw - mxp) : 0.f;
  const float en = act ? expf(-raw - mxn) : 0.f;

  red[tid] = ep;
  __syncthreads();
  if (tid < 32) red[tid] += red[tid + 128];
  __syncthreads();
  for (int s = 64; s > 0; s >>= 1) { if (tid < s) red[tid] += red[tid + s]; __syncthreads(); }
  const float sump = red[0];
  __syncthreads();
  red[tid] = en;
  __syncthreads();
  if (tid < 32) red[tid] += red[tid + 128];
  __syncthreads();
  for (int s = 64; s > 0; s >>= 1) { if (tid < s) red[tid] += red[tid + s]; __syncthreads(); }
  const float sumn = red[0];

  const long o = ((long)bh * SP + i) * SP + tid;
  pa[o] = (_Float16)(en / sumn);             // softmax(-scores)
  pc[o] = (_Float16)(ep / sump);             // softmax(+scores)
}

// ---------------------------------------------------------------------------
// Batched P*V GEMM per (b,h): o[b, m, h*64+d] = sum_j P[m,j] * V[b, j, h*64+d]
// ---------------------------------------------------------------------------
__global__ void av_wmma(const _Float16* __restrict__ p,
                        const _Float16* __restrict__ v,
                        _Float16* __restrict__ o) {
  const int lane = threadIdx.x;
  const int bh = blockIdx.y, b = bh / NH, h = bh % NH;
  const int mo = blockIdx.x << 4;            // 10 row tiles
  const _Float16* pb = p + (long)bh * SP * SP;
  const _Float16* vb = v + (long)b * S_ * D_ + h * DK;
  const int arow = imin(mo + (lane & 15), S_ - 1);
  const int kh = (lane >> 4) << 3;
  v8f acc[4] = {};
  for (int k0 = 0; k0 < SP; k0 += 32) {
    v16h a = load_frag_k(pb + (long)arow * SP + k0 + kh);
#pragma unroll
    for (int t = 0; t < 4; ++t) {
      const int n = t * 16 + (lane & 15);    // d within head, always < 64
      v16h bf;
#pragma unroll
      for (int i = 0; i < 8; ++i) {
        const int j0 = imin(k0 + kh + i, S_ - 1);        // P pad is zero -> safe clamp
        const int j1 = imin(k0 + 16 + kh + i, S_ - 1);
        bf[i]     = vb[(long)j0 * D_ + n];
        bf[8 + i] = vb[(long)j1 * D_ + n];
      }
      acc[t] = wmma_f16(a, bf, acc[t]);
    }
  }
  const int mbase = mo + ((lane >> 4) << 3);
#pragma unroll
  for (int r = 0; r < 8; ++r) {
    const int m = mbase + r;
    if (m >= S_) continue;
#pragma unroll
    for (int t = 0; t < 4; ++t) {
      const int n = t * 16 + (lane & 15);
      o[((long)b * S_ + m) * D_ + h * DK + n] = (_Float16)acc[t][r];
    }
  }
}

// out[b, s-1, :] = x1[b, s, :] + xc[b, s, :], s = 1..S_-1
__global__ void final_add_kernel(const float* __restrict__ x1,
                                 const float* __restrict__ xc,
                                 float* __restrict__ out) {
  const long i = (long)blockIdx.x * blockDim.x + threadIdx.x;
  const long total = (long)B_ * (S_ - 1) * D_;
  if (i >= total) return;
  const int b = (int)(i / ((S_ - 1) * D_));
  const int r = (int)(i % ((S_ - 1) * D_));
  const int s = r / D_ + 1;
  const int d = r % D_;
  const long src = ((long)b * S_ + s) * D_ + d;
  out[i] = x1[src] + xc[src];
}

// ---------------------------------------------------------------------------
// Host orchestration
// ---------------------------------------------------------------------------
extern "C" void kernel_launch(void* const* d_in, const int* in_sizes, int n_in,
                              void* d_out, int out_size, void* d_ws, size_t ws_size,
                              hipStream_t stream) {
  (void)in_sizes; (void)n_in; (void)out_size; (void)ws_size;
  const float* x    = (const float*)d_in[0];
  const float* attn = (const float*)d_in[1];
  const float* Wq = (const float*)d_in[2];  const float* bq = (const float*)d_in[3];
  const float* Wk = (const float*)d_in[4];  const float* bk = (const float*)d_in[5];
  const float* Wv = (const float*)d_in[6];  const float* bv = (const float*)d_in[7];
  const float* Wo = (const float*)d_in[8];  const float* bo = (const float*)d_in[9];
  const float* ln1g = (const float*)d_in[10]; const float* ln1b = (const float*)d_in[11];
  const float* ln2g = (const float*)d_in[12]; const float* ln2b = (const float*)d_in[13];
  const float* W1 = (const float*)d_in[14]; const float* b1 = (const float*)d_in[15];
  const float* W2 = (const float*)d_in[16]; const float* b2 = (const float*)d_in[17];
  float* out = (float*)d_out;

  // Workspace carve (256B aligned slices)
  char* wp = (char*)d_ws;
  auto carve = [&](size_t bytes) { char* p = wp; wp += (bytes + 255) & ~(size_t)255; return p; };
  const int BH = B_ * NH;                       // 96
  float*     v0   = (float*)carve(sizeof(float) * BH * N_);
  float*     v1   = (float*)carve(sizeof(float) * BH * N_);
  int*       idx  = (int*)carve(sizeof(int) * B_ * NH * KSEL);
  float*     fl   = (float*)carve(sizeof(float) * MROWS * D_);
  _Float16*  n1h  = (_Float16*)carve(sizeof(_Float16) * MROWS * D_);
  _Float16*  WqT  = (_Float16*)carve(sizeof(_Float16) * D_ * D_);
  _Float16*  WkT  = (_Float16*)carve(sizeof(_Float16) * D_ * D_);
  _Float16*  WvT  = (_Float16*)carve(sizeof(_Float16) * D_ * D_);
  _Float16*  WoT  = (_Float16*)carve(sizeof(_Float16) * D_ * D_);
  _Float16*  W1T  = (_Float16*)carve(sizeof(_Float16) * D_ * FF);
  _Float16*  W2T  = (_Float16*)carve(sizeof(_Float16) * FF * D_);
  _Float16*  qh   = (_Float16*)carve(sizeof(_Float16) * MROWS * D_);
  _Float16*  kh   = (_Float16*)carve(sizeof(_Float16) * MROWS * D_);
  _Float16*  vh   = (_Float16*)carve(sizeof(_Float16) * MROWS * D_);
  float*     sc   = (float*)carve(sizeof(float) * BH * SP * SP);
  _Float16*  pa   = (_Float16*)carve(sizeof(_Float16) * BH * SP * SP);
  _Float16*  pc   = (_Float16*)carve(sizeof(_Float16) * BH * SP * SP);
  _Float16*  o1h  = (_Float16*)carve(sizeof(_Float16) * MROWS * D_);
  _Float16*  o2h  = (_Float16*)carve(sizeof(_Float16) * MROWS * D_);
  float*     x1   = (float*)carve(sizeof(float) * MROWS * D_);
  float*     xc   = (float*)carve(sizeof(float) * MROWS * D_);
  _Float16*  lnh  = (_Float16*)carve(sizeof(_Float16) * MROWS * D_);
  _Float16*  h1h  = (_Float16*)carve(sizeof(_Float16) * MROWS * FF);

  // --- 1. weight conversion (f32 -> transposed f16) -------------------------
  {
    int g = (D_ * D_ + 255) / 256;
    convert_transpose_kernel<<<g, 256, 0, stream>>>(Wq, WqT, D_, D_);
    convert_transpose_kernel<<<g, 256, 0, stream>>>(Wk, WkT, D_, D_);
    convert_transpose_kernel<<<g, 256, 0, stream>>>(Wv, WvT, D_, D_);
    convert_transpose_kernel<<<g, 256, 0, stream>>>(Wo, WoT, D_, D_);
    int g2 = (D_ * FF + 255) / 256;
    convert_transpose_kernel<<<g2, 256, 0, stream>>>(W1, W1T, D_, FF);   // W1T[FF][D]
    convert_transpose_kernel<<<g2, 256, 0, stream>>>(W2, W2T, FF, D_);   // W2T[D][FF]
  }

  // --- 2. attention rollout (HBM-bandwidth dominated) -----------------------
  {
    dim3 grid((N_ + 255) / 256, BH);
    const float* attn3row0 = attn + (size_t)3 * BH * N_ * N_;  // attn[3][b][h][0][:]
    rollout_kernel<<<grid, 256, 0, stream>>>(attn3row0, (long)N_ * N_,
                                             attn + (size_t)2 * BH * N_ * N_, v0);
    rollout_kernel<<<grid, 256, 0, stream>>>(v0, (long)N_,
                                             attn + (size_t)1 * BH * N_ * N_, v1);
    rollout_kernel<<<grid, 256, 0, stream>>>(v1, (long)N_, attn, v0);
  }

  // --- 3. top-k token selection + gather + LN1 ------------------------------
  topk_kernel<<<BH, 192, 0, stream>>>(v0, idx);
  gather_ln_kernel<<<MROWS, 256, 0, stream>>>(x, idx, ln1g, ln1b, fl, n1h);

  // --- generic GEMM launcher (block = 128 M-rows x 64 N-cols, 8 waves) ------
  auto gemm = [&](const _Float16* A, int lda, const _Float16* Bt, int ldb,
                  const float* bias, const float* resid,
                  float* Cf, _Float16* Ch, int M, int N, int K, int ldc, int relu) {
    int blocks = ((M + 127) / 128) * (N / 64);
    gemm_bt_wmma<<<blocks, 256, 0, stream>>>(A, lda, Bt, ldb, bias, resid,
                                             Cf, Ch, M, N, K, ldc, relu);
  };

  // --- 4. QKV projections (WMMA + TDM-staged weights) -----------------------
  gemm(n1h, D_, WqT, D_, bq, nullptr, nullptr, qh, MROWS, D_, D_, D_, 0);
  gemm(n1h, D_, WkT, D_, bk, nullptr, nullptr, kh, MROWS, D_, D_, D_, 0);
  gemm(n1h, D_, WvT, D_, bv, nullptr, nullptr, vh, MROWS, D_, D_, D_, 0);

  // --- 5. scores, dual softmax, P*V (WMMA) ----------------------------------
  scores_wmma<<<dim3(100, BH), 32, 0, stream>>>(qh, kh, sc);
  softmax_pm_kernel<<<BH * S_, SP, 0, stream>>>(sc, pa, pc);
  av_wmma<<<dim3(10, BH), 32, 0, stream>>>(pa, vh, o1h);
  av_wmma<<<dim3(10, BH), 32, 0, stream>>>(pc, vh, o2h);

  // --- 6. output projections: x1 = fl + o1@Wo + bo ; xc = o2@Wo + bo --------
  gemm(o1h, D_, WoT, D_, bo, fl, x1, nullptr, MROWS, D_, D_, D_, 0);
  gemm(o2h, D_, WoT, D_, bo, nullptr, xc, nullptr, MROWS, D_, D_, D_, 0);

  // --- 7. FFN sublayer on x1 (in-place residual) ----------------------------
  ln_kernel<<<MROWS, 256, 0, stream>>>(x1, ln2g, ln2b, lnh);
  gemm(lnh, D_, W1T, D_, b1, nullptr, nullptr, h1h, MROWS, FF, D_, FF, 1);
  gemm(h1h, FF, W2T, FF, b2, x1, x1, nullptr, MROWS, D_, FF, D_, 0);

  // --- 8. FFN sublayer on xc (in-place residual) ----------------------------
  ln_kernel<<<MROWS, 256, 0, stream>>>(xc, ln2g, ln2b, lnh);
  gemm(lnh, D_, W1T, D_, b1, nullptr, nullptr, h1h, MROWS, FF, D_, FF, 1);
  gemm(h1h, FF, W2T, FF, b2, xc, xc, nullptr, MROWS, D_, FF, D_, 0);

  // --- 9. out = (x1 + xc)[:, 1:] --------------------------------------------
  {
    long total = (long)B_ * (S_ - 1) * D_;
    final_add_kernel<<<(int)((total + 255) / 256), 256, 0, stream>>>(x1, xc, out);
  }
}
